// PAM_71562745086361
// MI455X (gfx1250) — matrix operand order repaired and checked by reading, not compiled, not run on hardware
//
#include <hip/hip_runtime.h>

typedef _Float16 half_t;
typedef __attribute__((ext_vector_type(16))) _Float16 v16h;
typedef __attribute__((ext_vector_type(8)))  _Float16 v8h;
typedef __attribute__((ext_vector_type(8)))  float    v8f;
typedef __attribute__((ext_vector_type(4)))  float    v4f;
typedef __attribute__((ext_vector_type(4)))  int      v4i;
typedef __attribute__((address_space(1))) v4i as1_v4i;  // global int4
typedef __attribute__((address_space(3))) v4i as3_v4i;  // LDS int4

#define NB 4
#define CC 512
#define DD 64
#define NN 4096

// Async global->LDS staging (GLOBAL_LOAD_ASYNC_TO_LDS_B128, ASYNCcnt) when the
// toolchain exposes the builtin; otherwise fall back to VGPR-bounced copies.
#if defined(__gfx1250__) && \
    __has_builtin(__builtin_amdgcn_global_load_async_to_lds_b128)
#define HAVE_ASYNC_LDS 1
#else
#define HAVE_ASYNC_LDS 0
#endif

static __device__ __forceinline__ v8f wmma_f16(v16h a, v16h b, v8f c) {
  // D = A(16x32 f16) * B(32x16 f16) + C(16x16 f32)
  return __builtin_amdgcn_wmma_f32_16x16x32_f16(false, a, false, b, (short)0, c,
                                                false, false);
}

// 16 contiguous halves -> v16h (two aligned 16B loads, no scalar packing).
static __device__ __forceinline__ v16h ld16(const half_t* p) {
  v8h lo = *(const v8h*)p;
  v8h hi = *(const v8h*)(p + 8);
  return __builtin_shufflevector(lo, hi, 0, 1, 2, 3, 4, 5, 6, 7, 8, 9, 10, 11,
                                 12, 13, 14, 15);
}

// A fragment: 16x32 f16 tile from row-major src[m][k].
// lane m = lane&15, element s -> k = 16*(lane>>4) + s (matches B fragments)
// => per lane: 16 consecutive halves [k0+16h, +16): two aligned b128 loads.
static __device__ __forceinline__ v16h load_a(const half_t* s, int stride,
                                              int m0, int k0, int lane) {
  return ld16(s + (m0 + (lane & 15)) * stride + k0 + ((lane >> 4) << 4));
}

// B fragment: B[k][n] with the source tile stored n-major: src[n][k].
// lane n = lane&15, element s -> k = 16*(lane>>4) + s (same order as A)
// => per lane: 16 consecutive halves of row n starting at k0 + 16h.
static __device__ __forceinline__ v16h load_bT(const half_t* s, int stride,
                                               int k0, int n0, int lane) {
  return ld16(s + (n0 + (lane & 15)) * stride + k0 + ((lane >> 4) << 4));
}

// Y = x^T W^T + bias.
// n_major=1: Y[b][n][OUT]  (K buffer, row per position, OUT=64)
// n_major=0: Y[b][OUT][NN] (V buffer, transposed: row per channel)
// grid: (NN/64, OUT/64, NB), block 256 (8 waves), 64x64 output tile / block.
__global__ __launch_bounds__(256) void pam_proj(const float* __restrict__ x,
                                                const float* __restrict__ W,
                                                const float* __restrict__ bias,
                                                half_t* __restrict__ Y,
                                                int OUT, int n_major) {
  __shared__ __align__(16) half_t lds_a[64][40];  // A tile: 64 n x 32 c
  __shared__ __align__(16) half_t lds_b[64][40];  // B tile: 64 o x 32 c
  __shared__ __align__(16) half_t lds_c[64][72];  // output bounce: [n][o]
  const int t = threadIdx.x;
  const int lane = t & 31;
  const int w = t >> 5;
  const int h = lane >> 4;
  const int b = blockIdx.z;
  const int n0 = blockIdx.x * 64;
  const int o0 = blockIdx.y * 64;
  const int mt = w & 3;          // m tile 0..3
  const int ntp = (w >> 2) * 2;  // n tile pair base (0 or 2)
  const int row = t & 63;        // staging row owned by this thread
  const int cg = t >> 6;         // staging c-group (8 elements)

  const v8f zero = {0.f, 0.f, 0.f, 0.f, 0.f, 0.f, 0.f, 0.f};
  v8f acc[2] = {zero, zero};

  for (int kc = 0; kc < CC; kc += 32) {
    __syncthreads();
    // Stage A: 8 coalesced b32 loads (lanes sweep contiguous n), pack to one
    // 16B LDS store per thread.
    {
      v8h pk;
#pragma unroll
      for (int cc = 0; cc < 8; ++cc)
        pk[cc] =
            (half_t)x[((size_t)b * CC + kc + cg * 8 + cc) * NN + n0 + row];
      *(v8h*)&lds_a[row][cg * 8] = pk;
    }
    // Stage B: W rows are contiguous -> two 16B vector loads, one 16B store.
    {
      const float* wp = &W[(size_t)(o0 + row) * CC + kc + cg * 8];
      v4f w0 = *(const v4f*)wp;
      v4f w1 = *(const v4f*)(wp + 4);
      v8h pk;
#pragma unroll
      for (int cc = 0; cc < 4; ++cc) {
        pk[cc] = (half_t)w0[cc];
        pk[4 + cc] = (half_t)w1[cc];
      }
      *(v8h*)&lds_b[row][cg * 8] = pk;
    }
    __syncthreads();
    v16h a = load_a(&lds_a[0][0], 40, mt * 16, 0, lane);
#pragma unroll
    for (int j = 0; j < 2; ++j) {
      v16h bb = load_bT(&lds_b[0][0], 40, 0, (ntp + j) * 16, lane);
      acc[j] = wmma_f16(a, bb, acc[j]);
    }
  }

  if (n_major) {
    // Transpose through LDS (cheap scalar b16 on the LDS side), then
    // coalesced 16B global stores per thread.
#pragma unroll
    for (int j = 0; j < 2; ++j) {
      const int oc = (ntp + j) * 16 + (lane & 15);
      const float bv = bias[o0 + oc];
#pragma unroll
      for (int v = 0; v < 8; ++v)
        lds_c[mt * 16 + 8 * h + v][oc] = (half_t)(acc[j][v] + bv);
    }
    __syncthreads();
#pragma unroll
    for (int u = 0; u < 2; ++u) {
      const int ch = u * 256 + t;
      const int r = ch >> 3;
      const int sg = ch & 7;
      *(v8h*)&Y[((size_t)b * NN + n0 + r) * OUT + o0 + sg * 8] =
          *(const v8h*)&lds_c[r][sg * 8];
    }
  } else {
    // V^T layout: each lane's 8 rows are contiguous n -> direct 16B stores.
#pragma unroll
    for (int j = 0; j < 2; ++j) {
      const int oc = o0 + (ntp + j) * 16 + (lane & 15);
      const float bv = bias[oc];
      v8h pk;
#pragma unroll
      for (int v = 0; v < 8; ++v) pk[v] = (half_t)(acc[j][v] + bv);
      *(v8h*)&Y[((size_t)b * OUT + oc) * NN + n0 + mt * 16 + 8 * h] = pk;
    }
  }
}

// Stage one 16B chunk of a 64x64 fp16 tile into LDS (async when available).
static __device__ __forceinline__ void stage_tile_chunk(
    half_t* lds_row, const half_t* gsrc) {
#if HAVE_ASYNC_LDS
  __builtin_amdgcn_global_load_async_to_lds_b128((as1_v4i*)gsrc,
                                                 (as3_v4i*)lds_row, 0, 0);
#else
  *(v8h*)lds_row = *(const v8h*)gsrc;
#endif
}

static __device__ __forceinline__ void stage_wait() {
#if HAVE_ASYNC_LDS
  asm volatile("s_wait_asynccnt 0" ::: "memory");
#endif
}

// Fused flash-style attention + residual:
// out[b,c,i] = gamma * sum_j softmax_j(K[i]·K[j]) * V[c,j] + x[b,c,i]
// grid: (NN/64, NB), block 256 (8 waves). 64 queries per block; each wave owns
// a 64-channel output slice (16 f32 accum tiles = 128 VGPRs/lane).
__global__ __launch_bounds__(256) void pam_attn(const float* __restrict__ x,
                                                const half_t* __restrict__ Kbuf,
                                                const half_t* __restrict__ Vbuf,
                                                const float* __restrict__ gamma_p,
                                                float* __restrict__ out) {
  __shared__ __align__(16) half_t lds_q[64][72];
  __shared__ __align__(16) half_t lds_k[64][72];
  __shared__ __align__(16) float  lds_st[64][68];  // S transposed: [key][query]
  __shared__ __align__(16) half_t lds_p[64][72];   // P row-major: [query][key]
  __shared__ __align__(16) float  m_row[64];
  __shared__ __align__(16) float  l_row[64];
  __shared__ __align__(16) float  scale_row[64];
  __shared__ __align__(16) float  part_max[64][4];
  __shared__ __align__(16) float  part_sum[64][4];

  const int t = threadIdx.x;
  const int lane = t & 31;
  const int w = t >> 5;
  const int h = lane >> 4;
  const int b = blockIdx.y;
  const int q0 = blockIdx.x * 64;
  const float gamma = gamma_p[0];

  const half_t* Kb = Kbuf + (size_t)b * NN * DD;  // [N][64]
  const half_t* Vb = Vbuf + (size_t)b * CC * NN;  // [C][N] (transposed)

  // Load Q block (64x64 fp16): 512 x 16B chunks, 2 per thread.
#pragma unroll
  for (int u = 0; u < 2; ++u) {
    const int ch = u * 256 + t;
    const int r = ch >> 3;
    const int sg = ch & 7;
    stage_tile_chunk(&lds_q[r][sg * 8], &Kb[(size_t)(q0 + r) * DD + sg * 8]);
  }
  if (t < 64) { m_row[t] = -INFINITY; l_row[t] = 0.f; }

  const v8f zero = {0.f, 0.f, 0.f, 0.f, 0.f, 0.f, 0.f, 0.f};
  v8f acc_o[16];
#pragma unroll
  for (int i = 0; i < 16; ++i) acc_o[i] = zero;

  const int smt = w >> 1;       // S tile row for this wave
  const int snt = (w & 1) * 2;  // S tile col pair base
  const int cbase = w * 64;     // channel slice owned by this wave

  for (int jb = 0; jb < NN; jb += 64) {
    __syncthreads();
    // Stage K block (async b128 direct to LDS) + prefetch next K/V blocks.
#pragma unroll
    for (int u = 0; u < 2; ++u) {
      const int ch = u * 256 + t;
      const int r = ch >> 3;
      const int sg = ch & 7;
      stage_tile_chunk(&lds_k[r][sg * 8], &Kb[(size_t)(jb + r) * DD + sg * 8]);
    }
    if (jb + 64 < NN) {
      __builtin_prefetch(&Kb[(size_t)(jb + 64) * DD + t * 16], 0, 0);
      const half_t* vpf = Vb + (size_t)(cbase + lane * 2) * NN + jb + 64;
      __builtin_prefetch(vpf, 0, 0);
      __builtin_prefetch(vpf + NN, 0, 0);
    }
    stage_wait();
    __syncthreads();

    // S = Q * K^T : 16 tiles / 8 waves = 2 per wave, K-dim = 64.
    v8f s0 = zero, s1 = zero;
#pragma unroll
    for (int ks = 0; ks < 2; ++ks) {
      v16h a = load_a(&lds_q[0][0], 72, smt * 16, ks * 32, lane);
      v16h b0 = load_bT(&lds_k[0][0], 72, ks * 32, snt * 16, lane);
      v16h b1 = load_bT(&lds_k[0][0], 72, ks * 32, (snt + 1) * 16, lane);
      s0 = wmma_f16(a, b0, s0);
      s1 = wmma_f16(a, b1, s1);
    }
    // Store S transposed: lane's 8 rows are consecutive q -> 2x v4f per tile.
    {
      const int rb = smt * 16 + 8 * h;
      const int c0 = snt * 16 + (lane & 15);
      *(v4f*)&lds_st[c0][rb] = __builtin_shufflevector(s0, s0, 0, 1, 2, 3);
      *(v4f*)&lds_st[c0][rb + 4] = __builtin_shufflevector(s0, s0, 4, 5, 6, 7);
      *(v4f*)&lds_st[c0 + 16][rb] = __builtin_shufflevector(s1, s1, 0, 1, 2, 3);
      *(v4f*)&lds_st[c0 + 16][rb + 4] =
          __builtin_shufflevector(s1, s1, 4, 5, 6, 7);
    }
    __syncthreads();

    // Online softmax, 4 threads per row (16 cols each).
    {
      const int r = t >> 2, qd = t & 3;
      float pm = -INFINITY;
#pragma unroll
      for (int j = 0; j < 16; ++j) pm = fmaxf(pm, lds_st[qd * 16 + j][r]);
      part_max[r][qd] = pm;
      __syncthreads();
      float mmax = fmaxf(fmaxf(part_max[r][0], part_max[r][1]),
                         fmaxf(part_max[r][2], part_max[r][3]));
      mmax = fmaxf(mmax, m_row[r]);
      float sum = 0.f;
      float pv[16];
#pragma unroll
      for (int j = 0; j < 16; ++j) {
        pv[j] = __expf(lds_st[qd * 16 + j][r] - mmax);
        sum += pv[j];
      }
      v8h p0, p1;
#pragma unroll
      for (int j = 0; j < 8; ++j) {
        p0[j] = (half_t)pv[j];
        p1[j] = (half_t)pv[8 + j];
      }
      *(v8h*)&lds_p[r][qd * 16] = p0;
      *(v8h*)&lds_p[r][qd * 16 + 8] = p1;
      part_sum[r][qd] = sum;
      __syncthreads();
      if (qd == 0) {
        const float sc = __expf(m_row[r] - mmax);  // 0 on first block
        l_row[r] = l_row[r] * sc + part_sum[r][0] + part_sum[r][1] +
                   part_sum[r][2] + part_sum[r][3];
        m_row[r] = mmax;
        scale_row[r] = sc;
      }
    }
    __syncthreads();

    // Rescale running output by exp(m_old - m_new): vectorized LDS reads.
    v4f sc0[4], sc1[4];
#pragma unroll
    for (int mt = 0; mt < 4; ++mt) {
      sc0[mt] = *(const v4f*)&scale_row[mt * 16 + 8 * h];
      sc1[mt] = *(const v4f*)&scale_row[mt * 16 + 8 * h + 4];
    }
#pragma unroll
    for (int i = 0; i < 16; ++i) {
      const int mt = i >> 2;
#pragma unroll
      for (int v = 0; v < 4; ++v) {
        acc_o[i][v] *= sc0[mt][v];
        acc_o[i][4 + v] *= sc1[mt][v];
      }
    }

    // O += P * V. V is [C][N]: per-lane fragment = 32 contiguous bytes.
#pragma unroll
    for (int ks = 0; ks < 2; ++ks) {
      v16h bf[4];
#pragma unroll
      for (int nt = 0; nt < 4; ++nt) {
        const int c = cbase + nt * 16 + (lane & 15);
        bf[nt] = ld16(Vb + (size_t)c * NN + jb + ks * 32 + h * 16);
      }
#pragma unroll
      for (int mt = 0; mt < 4; ++mt) {
        v16h a = load_a(&lds_p[0][0], 72, mt * 16, ks * 32, lane);
#pragma unroll
        for (int nt = 0; nt < 4; ++nt)
          acc_o[mt * 4 + nt] = wmma_f16(a, bf[nt], acc_o[mt * 4 + nt]);
      }
    }
  }
  __syncthreads();

  // Invert l once per row (64 rcp) so the epilogue is pure mul/fma.
  if (t < 64) l_row[t] = 1.f / l_row[t];
  __syncthreads();

  // Epilogue: out = gamma * O * (1/l) + x (two 16B f32 vector ops per tile).
#pragma unroll
  for (int i = 0; i < 16; ++i) {
    const int mt = i >> 2, nt = i & 3;
    const int c = cbase + nt * 16 + (lane & 15);
    const int rbase = mt * 16 + 8 * h;  // 8 consecutive query rows per lane
    const size_t base = ((size_t)b * CC + c) * NN + q0 + rbase;
#pragma unroll
    for (int g = 0; g < 2; ++g) {
      v4f xv = *(const v4f*)&x[base + 4 * g];
      v4f lv = *(const v4f*)&l_row[rbase + 4 * g];
      v4f ov;
#pragma unroll
      for (int v = 0; v < 4; ++v)
        ov[v] = gamma * (acc_o[i][4 * g + v] * lv[v]) + xv[v];
      *(v4f*)&out[base + 4 * g] = ov;
    }
  }
}

extern "C" void kernel_launch(void* const* d_in, const int* in_sizes, int n_in,
                              void* d_out, int out_size, void* d_ws, size_t ws_size,
                              hipStream_t stream) {
  const float* x     = (const float*)d_in[0];
  const float* Wq    = (const float*)d_in[1];
  const float* bq    = (const float*)d_in[2];
  const float* Wv    = (const float*)d_in[3];
  const float* bv    = (const float*)d_in[4];
  const float* gamma = (const float*)d_in[5];
  float* out = (float*)d_out;

  // Workspace: K fp16 [B,N,64] = 2 MB, V^T fp16 [B,C,N] = 16 MB.
  half_t* Kbuf = (half_t*)d_ws;
  half_t* Vbuf = Kbuf + (size_t)NB * NN * DD;

  dim3 gk(NN / 64, DD / 64, NB);  // 64x1x4
  pam_proj<<<gk, 256, 0, stream>>>(x, Wq, bq, Kbuf, DD, 1);

  dim3 gv(NN / 64, CC / 64, NB);  // 64x8x4
  pam_proj<<<gv, 256, 0, stream>>>(x, Wv, bv, Vbuf, CC, 0);

  dim3 ga(NN / 64, NB);           // 64x4
  pam_attn<<<ga, 256, 0, stream>>>(x, Kbuf, Vbuf, gamma, out);
}
